// Attention_32856499814873
// MI455X (gfx1250) — matrix operand order repaired and checked by reading, not compiled, not run on hardware
//
#include <hip/hip_runtime.h>

#define T_SEQ 4096
#define C_DIM 1024
#define H_NUM 16
#define DH    64

typedef __attribute__((ext_vector_type(16))) __bf16 v16bf;
typedef __attribute__((ext_vector_type(8)))  __bf16 v8bf;
typedef __attribute__((ext_vector_type(8)))  float  v8f;

union V16 { v16bf v; v8bf h[2]; };

__device__ __forceinline__ v8bf ld8(const __bf16* p) { return *(const v8bf*)p; }

__device__ __forceinline__ v8f zero8() {
  v8f z = {0.f, 0.f, 0.f, 0.f, 0.f, 0.f, 0.f, 0.f};
  return z;
}

// ---------------- fp32 -> bf16 elementwise convert ----------------
__global__ void cvt_bf16(const float* __restrict__ in, __bf16* __restrict__ out, int n) {
  for (int i = blockIdx.x * blockDim.x + threadIdx.x; i < n; i += gridDim.x * blockDim.x)
    out[i] = (__bf16)in[i];
}

// ---------------- Wt[k][n] = (bf16) W[n][k]  (DIM x DIM) ----------------
__global__ void transpose_cvt(const float* __restrict__ W, __bf16* __restrict__ Wt) {
  int k = blockIdx.x * 16 + threadIdx.x;
  int n = blockIdx.y * 16 + threadIdx.y;
  Wt[(size_t)k * C_DIM + n] = (__bf16)W[(size_t)n * C_DIM + k];
}

// ---------------- WMMA GEMM: out = A[M,K] * B[K,N] + bias ----------------
// A row-major bf16, B row-major bf16 (B = W^T already), bias fp32 per column.
// Each wave: 32x64 tile (2 m-tiles x 4 n-tiles) -> every B fragment feeds two
// WMMAs, halving B-fragment issue per FLOP. 8 waves/block -> 256x64 block tile.
template<bool TRANS_OUT, bool OUT_F32>
__global__ __launch_bounds__(256) void wmma_gemm(const __bf16* __restrict__ A,
                                                 const __bf16* __restrict__ B,
                                                 const float* __restrict__ bias,
                                                 void* __restrict__ outp,
                                                 int M, int N, int K) {
  const int lane = threadIdx.x & 31;
  const int wav  = threadIdx.x >> 5;
  const int m0   = blockIdx.y * 256 + wav * 32;
  const int n0   = blockIdx.x * 64;
  const int m    = lane & 15;
  const int hi   = lane >> 4;

  v8f acc[2][4];
#pragma unroll
  for (int mt = 0; mt < 2; ++mt)
#pragma unroll
    for (int nt = 0; nt < 4; ++nt) acc[mt][nt] = zero8();

  const __bf16* arow0 = A + (size_t)(m0 + m) * K;
  const __bf16* arow1 = A + (size_t)(m0 + 16 + m) * K;
  for (int k0 = 0; k0 < K; k0 += 32) {
    V16 a0, a1;
    a0.h[0] = ld8(arow0 + k0 + 8 * hi);        // K = 8*hi .. 8*hi+7
    a0.h[1] = ld8(arow0 + k0 + 16 + 8 * hi);   // K = 16+8*hi .. 23+8*hi
    a1.h[0] = ld8(arow1 + k0 + 8 * hi);
    a1.h[1] = ld8(arow1 + k0 + 16 + 8 * hi);
    const __bf16* brow = B + (size_t)k0 * N + n0 + m;  // lane%16 = N column of B, elements = K rows
    // speculative prefetch of next k-block (global_prefetch_b8; OOB is dropped)
    __builtin_prefetch(arow0 + k0 + 32, 0, 3);
    __builtin_prefetch(brow + (size_t)32 * N, 0, 3);
#pragma unroll
    for (int nt = 0; nt < 4; ++nt) {
      V16 b;
#pragma unroll
      for (int s = 0; s < 16; ++s) b.v[s] = brow[(size_t)(s < 8 ? 8 * hi + s : 8 + 8 * hi + s) * N + nt * 16];
      acc[0][nt] = __builtin_amdgcn_wmma_f32_16x16x32_bf16(
          false, a0.v, false, b.v, (short)0, acc[0][nt], false, false);
      acc[1][nt] = __builtin_amdgcn_wmma_f32_16x16x32_bf16(
          false, a1.v, false, b.v, (short)0, acc[1][nt], false, false);
    }
  }

#pragma unroll
  for (int mt = 0; mt < 2; ++mt) {
#pragma unroll
    for (int nt = 0; nt < 4; ++nt) {
      const int col = n0 + nt * 16 + m;
      const float bval = bias[col];
#pragma unroll
      for (int e = 0; e < 8; ++e) {
        const int row = m0 + mt * 16 + e + 8 * hi;   // C/D layout: M = e + 8*(lane/16)
        const float val = acc[mt][nt][e] + bval;
        if (OUT_F32)
          ((float*)outp)[(size_t)row * N + col] = val;
        else if (TRANS_OUT)
          ((__bf16*)outp)[(size_t)col * M + row] = (__bf16)val;   // Kt[c][t]
        else
          ((__bf16*)outp)[(size_t)row * N + col] = (__bf16)val;
      }
    }
  }
}

// ---------------- Flash attention ----------------
// Q  : [T][C] bf16 (per-head slice starts at h*DH in a row)
// Kt : [C][T] bf16 (head-transposed: row = feature, col = token)
// V  : [T][C] bf16
// O  : [T][C] bf16
// One wave per 16-query tile of one head; 8 waves / 256-thread block.
__global__ __launch_bounds__(256) void flash_attn(const __bf16* __restrict__ Q,
                                                  const __bf16* __restrict__ Kt,
                                                  const __bf16* __restrict__ V,
                                                  __bf16* __restrict__ O) {
  __shared__ __bf16 ldsP[8][16 * 32];   // per-wave P tile staging (1 KB each)

  const int lane = threadIdx.x & 31;
  const int wav  = threadIdx.x >> 5;
  const int h    = blockIdx.y;
  const int q0   = blockIdx.x * 128 + wav * 16;
  const int m    = lane & 15;
  const int hi   = lane >> 4;
  const float scale = 0.03125f;         // C_DIM^-0.5 (reference uses full-dim scale)
  const float NEG   = -1e30f;

  // Q tile 16x64 -> two 16x32 A-fragments (resident for whole KV loop)
  const __bf16* qrow = Q + (size_t)(q0 + m) * C_DIM + h * DH;
  V16 aq0, aq1;
  aq0.h[0] = ld8(qrow + 8 * hi);        aq0.h[1] = ld8(qrow + 16 + 8 * hi);
  aq1.h[0] = ld8(qrow + 32 + 8 * hi);   aq1.h[1] = ld8(qrow + 48 + 8 * hi);

  v8f o0 = zero8(), o1 = zero8(), o2 = zero8(), o3 = zero8();
  float mr[8], lr[8];
#pragma unroll
  for (int e = 0; e < 8; ++e) { mr[e] = NEG; lr[e] = 0.f; }

  for (int j0 = 0; j0 < q0 + 16; j0 += 32) {
    // ---- S = Q K^T : 16x32 as two 16x16 n-tiles, K(=d) reduced over 2 steps ----
    v8f s0 = zero8(), s1 = zero8();
    const __bf16* k0p = Kt + ((size_t)(h * DH)) * T_SEQ + j0 + m;       // d block 0
    const __bf16* k1p = Kt + ((size_t)(h * DH) + 32) * T_SEQ + j0 + m;  // d block 1
    V16 b;
    for (int s = 0; s < 16; ++s) b.v[s] = k0p[(size_t)(s < 8 ? 8 * hi + s : 8 + 8 * hi + s) * T_SEQ];
    s0 = __builtin_amdgcn_wmma_f32_16x16x32_bf16(false, aq0.v, false, b.v, (short)0, s0, false, false);
    for (int s = 0; s < 16; ++s) b.v[s] = k1p[(size_t)(s < 8 ? 8 * hi + s : 8 + 8 * hi + s) * T_SEQ];
    s0 = __builtin_amdgcn_wmma_f32_16x16x32_bf16(false, aq1.v, false, b.v, (short)0, s0, false, false);
    for (int s = 0; s < 16; ++s) b.v[s] = k0p[(size_t)(s < 8 ? 8 * hi + s : 8 + 8 * hi + s) * T_SEQ + 16];
    s1 = __builtin_amdgcn_wmma_f32_16x16x32_bf16(false, aq0.v, false, b.v, (short)0, s1, false, false);
    for (int s = 0; s < 16; ++s) b.v[s] = k1p[(size_t)(s < 8 ? 8 * hi + s : 8 + 8 * hi + s) * T_SEQ + 16];
    s1 = __builtin_amdgcn_wmma_f32_16x16x32_bf16(false, aq1.v, false, b.v, (short)0, s1, false, false);

    // ---- causal mask + scale + online softmax (f32, cross-lane within half) ----
    float sca[8];
#pragma unroll
    for (int e = 0; e < 8; ++e) {
      const int qg = q0 + e + 8 * hi;
      float v0 = s0[e] * scale; if (j0 + m      > qg) v0 = NEG;
      float v1 = s1[e] * scale; if (j0 + 16 + m > qg) v1 = NEG;
      float t = fmaxf(v0, v1);
      t = fmaxf(t, __shfl_xor(t, 1, 32));
      t = fmaxf(t, __shfl_xor(t, 2, 32));
      t = fmaxf(t, __shfl_xor(t, 4, 32));
      t = fmaxf(t, __shfl_xor(t, 8, 32));
      const float mnew = fmaxf(mr[e], t);
      const float sc   = __expf(mr[e] - mnew);
      const float p0   = __expf(v0 - mnew);
      const float p1   = __expf(v1 - mnew);
      float rs = p0 + p1;
      rs += __shfl_xor(rs, 1, 32);
      rs += __shfl_xor(rs, 2, 32);
      rs += __shfl_xor(rs, 4, 32);
      rs += __shfl_xor(rs, 8, 32);
      lr[e]  = lr[e] * sc + rs;
      mr[e]  = mnew;
      sca[e] = sc;
      const int row = e + 8 * hi;
      ldsP[wav][row * 32 + m]      = (__bf16)p0;
      ldsP[wav][row * 32 + 16 + m] = (__bf16)p1;
    }
    // wave-private LDS RAW: CDNA5 split-counter wait
    asm volatile("s_wait_dscnt 0" ::: "memory");

    // ---- reload P (16x32) in A-fragment layout ----
    V16 ap;
    const __bf16* prow = &ldsP[wav][m * 32];
    ap.h[0] = ld8(prow + 8 * hi);
    ap.h[1] = ld8(prow + 16 + 8 * hi);

    // ---- O = O*sc + P·V (d = 64 -> four 16x16 n-tiles) ----
#pragma unroll
    for (int e = 0; e < 8; ++e) {
      o0[e] *= sca[e]; o1[e] *= sca[e]; o2[e] *= sca[e]; o3[e] *= sca[e];
    }
    const __bf16* vrow = V + (size_t)j0 * C_DIM + h * DH + m;  // lane%16 = d column, elements = K(=t) rows
    for (int s = 0; s < 16; ++s) b.v[s] = vrow[(size_t)(s < 8 ? 8 * hi + s : 8 + 8 * hi + s) * C_DIM];
    o0 = __builtin_amdgcn_wmma_f32_16x16x32_bf16(false, ap.v, false, b.v, (short)0, o0, false, false);
    for (int s = 0; s < 16; ++s) b.v[s] = vrow[(size_t)(s < 8 ? 8 * hi + s : 8 + 8 * hi + s) * C_DIM + 16];
    o1 = __builtin_amdgcn_wmma_f32_16x16x32_bf16(false, ap.v, false, b.v, (short)0, o1, false, false);
    for (int s = 0; s < 16; ++s) b.v[s] = vrow[(size_t)(s < 8 ? 8 * hi + s : 8 + 8 * hi + s) * C_DIM + 32];
    o2 = __builtin_amdgcn_wmma_f32_16x16x32_bf16(false, ap.v, false, b.v, (short)0, o2, false, false);
    for (int s = 0; s < 16; ++s) b.v[s] = vrow[(size_t)(s < 8 ? 8 * hi + s : 8 + 8 * hi + s) * C_DIM + 48];
    o3 = __builtin_amdgcn_wmma_f32_16x16x32_bf16(false, ap.v, false, b.v, (short)0, o3, false, false);
  }

  // ---- normalize and store ----
#pragma unroll
  for (int e = 0; e < 8; ++e) {
    const float inv = 1.f / lr[e];
    const size_t base = (size_t)(q0 + e + 8 * hi) * C_DIM + h * DH + m;
    O[base]      = (__bf16)(o0[e] * inv);
    O[base + 16] = (__bf16)(o1[e] * inv);
    O[base + 32] = (__bf16)(o2[e] * inv);
    O[base + 48] = (__bf16)(o3[e] * inv);
  }
}

// ---------------- host-side orchestration ----------------
extern "C" void kernel_launch(void* const* d_in, const int* in_sizes, int n_in,
                              void* d_out, int out_size, void* d_ws, size_t ws_size,
                              hipStream_t stream) {
  (void)in_sizes; (void)n_in; (void)out_size; (void)ws_size;

  const float* x  = (const float*)d_in[0];
  const float* Wq = (const float*)d_in[1];
  const float* bq = (const float*)d_in[2];
  const float* Wk = (const float*)d_in[3];
  const float* bk = (const float*)d_in[4];
  const float* Wv = (const float*)d_in[5];
  const float* bv = (const float*)d_in[6];
  const float* Wp = (const float*)d_in[7];
  const float* bp = (const float*)d_in[8];
  float* out = (float*)d_out;

  const size_t TC = (size_t)T_SEQ * C_DIM;   // 4 Mi elems
  const size_t CC = (size_t)C_DIM * C_DIM;   // 1 Mi elems

  __bf16* p   = (__bf16*)d_ws;
  __bf16* xb  = p; p += TC;
  __bf16* wqt = p; p += CC;
  __bf16* wkt = p; p += CC;
  __bf16* wvt = p; p += CC;
  __bf16* wpt = p; p += CC;
  __bf16* qb  = p; p += TC;
  __bf16* ktb = p; p += TC;
  __bf16* vb  = p; p += TC;
  __bf16* aob = p; p += TC;   // total 48 MiB of workspace

  // 1) convert activations + weights to bf16 (weights transposed to [K][N])
  cvt_bf16<<<2048, 256, 0, stream>>>(x, xb, (int)TC);
  dim3 tg(C_DIM / 16, C_DIM / 16), tb(16, 16);
  transpose_cvt<<<tg, tb, 0, stream>>>(Wq, wqt);
  transpose_cvt<<<tg, tb, 0, stream>>>(Wk, wkt);
  transpose_cvt<<<tg, tb, 0, stream>>>(Wv, wvt);
  transpose_cvt<<<tg, tb, 0, stream>>>(Wp, wpt);

  // 2) QKV projections (K stored head-transposed Kt[c][t])
  dim3 gg(C_DIM / 64, T_SEQ / 256);
  wmma_gemm<false, false><<<gg, 256, 0, stream>>>(xb, wqt, bq, qb,  T_SEQ, C_DIM, C_DIM);
  wmma_gemm<true,  false><<<gg, 256, 0, stream>>>(xb, wkt, bk, ktb, T_SEQ, C_DIM, C_DIM);
  wmma_gemm<false, false><<<gg, 256, 0, stream>>>(xb, wvt, bv, vb,  T_SEQ, C_DIM, C_DIM);

  // 3) causal flash attention
  flash_attn<<<dim3(T_SEQ / 128, H_NUM), 256, 0, stream>>>(qb, ktb, vb, aob);

  // 4) output projection -> fp32 d_out
  wmma_gemm<false, true><<<gg, 256, 0, stream>>>(aob, wpt, bp, out, T_SEQ, C_DIM, C_DIM);
}
